// GraphHead_60653528154750
// MI455X (gfx1250) — matrix operand order repaired and checked by reading, not compiled, not run on hardware
//
#include <hip/hip_runtime.h>

// ---------------- CDNA5 WMMA types ----------------
typedef __attribute__((ext_vector_type(16))) __bf16 v16bf;
typedef __attribute__((ext_vector_type(8)))  float  v8f;

#define HDIM 128

// ---------------- fragment loaders ----------------
// A fragment: 16x32 bf16 from an f32 row-major matrix (ldA floats per row),
// converted on the fly. Contiguous-k layout used here (not AMD's table order):
//   lanes 0-15 : M=lane,    VGPR0-7 = K0..15
//   lanes16-31 : M=lane-16, VGPR0-7 = K16..31
__device__ __forceinline__ v16bf load_a_f32(const float* __restrict__ A, int ldA,
                                            int m0, int k0, int lane) {
  const int m  = m0 + (lane & 15);
  const int kb = k0 + ((lane & 16) ? 16 : 0);
  const float4* p = (const float4*)(A + (size_t)m * ldA + kb);
  float4 q0 = p[0];   // K kb+0..3
  float4 q1 = p[1];   // K kb+4..7
  float4 q2 = p[2];   // K kb+8..11
  float4 q3 = p[3];   // K kb+12..15
  v16bf a;
  a[0]=(__bf16)q0.x; a[1]=(__bf16)q0.y; a[2]=(__bf16)q0.z; a[3]=(__bf16)q0.w;
  a[4]=(__bf16)q1.x; a[5]=(__bf16)q1.y; a[6]=(__bf16)q1.z; a[7]=(__bf16)q1.w;
  a[8]=(__bf16)q2.x; a[9]=(__bf16)q2.y; a[10]=(__bf16)q2.z; a[11]=(__bf16)q2.w;
  a[12]=(__bf16)q3.x; a[13]=(__bf16)q3.y; a[14]=(__bf16)q3.z; a[15]=(__bf16)q3.w;
  return a;
}

// B fragment: 32x16 bf16 from a pre-transposed bf16 weight Wt[n][k] (ldB = K).
// Contiguous-k layout used here: lane holds column n = lane%16; K base 0 (lanes 0-15) / 16
// (lanes 16-31); VGPR j = packed pair (K=base+2j, base+2j+1) -> 16 contiguous
// bf16 = one 32-byte aligned load.
__device__ __forceinline__ v16bf load_b_bf16(const __bf16* __restrict__ Wt, int ldB,
                                             int n0, int k0, int lane) {
  const int n  = n0 + (lane & 15);
  const int kb = k0 + ((lane & 16) ? 16 : 0);
  return *(const v16bf*)(Wt + (size_t)n * ldB + kb);
}

// ---------------- small utility kernels ----------------
__global__ void k_zero(float* p, int n) {
  int i = blockIdx.x * blockDim.x + threadIdx.x;
  if (i < n) p[i] = 0.0f;
}

__global__ void k_count_deg(const int* __restrict__ row, const int* __restrict__ col,
                            float* __restrict__ deg, int E) {
  int e = blockIdx.x * blockDim.x + threadIdx.x;
  if (e >= E) return;
  if (row[e] != col[e]) atomicAdd(&deg[col[e]], 1.0f);
}

__global__ void k_deg_inv(float* __restrict__ deg, int n) {
  int i = blockIdx.x * blockDim.x + threadIdx.x;
  if (i < n) deg[i] = 1.0f / (deg[i] + 1.0f);   // segment count + self -> >= 1
}

__global__ void k_embed(const int* __restrict__ x, const float* __restrict__ emb,
                        float* __restrict__ z, int N) {
  long long gid = (long long)blockIdx.x * blockDim.x + threadIdx.x;
  int n = (int)(gid >> 7), f = (int)(gid & (HDIM - 1));
  if (n >= N) return;
  z[gid] = emb[(size_t)x[n] * HDIM + f];
}

// Wt[n*K + k] = (bf16) W[k*Nn + n]   (W is K x Nn row-major)
__global__ void k_wt_bf16(const float* __restrict__ W, __bf16* __restrict__ Wt,
                          int K, int Nn) {
  int idx = blockIdx.x * blockDim.x + threadIdx.x;
  if (idx >= K * Nn) return;
  int n = idx / K, k = idx % K;
  Wt[idx] = (__bf16)W[(size_t)k * Nn + n];
}

__global__ void k_init_agg(const float* __restrict__ deg_inv, const float* __restrict__ z,
                           float* __restrict__ agg, int N) {
  long long gid = (long long)blockIdx.x * blockDim.x + threadIdx.x;
  int n = (int)(gid >> 7);
  if (n >= N) return;
  agg[gid] = deg_inv[n] * z[gid];
}

// Message passing: agg[col] += deg_inv[col] * z[row]  (row != col)
// One 128-thread block per edge so row/col/deg_inv become *scalar* (SGPR)
// loads and the self-loop test a scalar branch; lanes do one coalesced f32
// gather of z[row] and one coalesced global_atomic_add_f32 each.
__global__ void __launch_bounds__(HDIM)
k_scatter(const int* __restrict__ row, const int* __restrict__ col,
          const float* __restrict__ deg_inv, const float* __restrict__ z,
          float* __restrict__ agg) {
  const int e = blockIdx.x;          // scalar
  const int f = threadIdx.x;         // 0..127
  const int r = row[e];              // uniform -> s_load
  const int c = col[e];              // uniform -> s_load
  if (r == c) return;                // scalar branch
  const float w = deg_inv[c];        // uniform
  atomicAdd(&agg[(size_t)c * HDIM + f], w * z[(size_t)r * HDIM + f]);
}

// ---------------- WMMA GEMM kernels ----------------
// z_out = relu(agg @ Wout + z @ Wroot + b). One wave per 16x128 OUTPUT STRIP:
// 8 f32 accumulators (64 VGPRs); each A fragment is loaded ONCE and reused by
// all 8 B tiles (weights stay L2/L0 resident) -> 8x less A traffic than
// tile-per-wave, 64 v_wmma_f32_16x16x32_bf16 per wave.
__global__ void k_gcn_gemm(const float* __restrict__ agg, const float* __restrict__ z,
                           const __bf16* __restrict__ WtOut, const __bf16* __restrict__ WtRoot,
                           const float* __restrict__ bias, float* __restrict__ zout,
                           int numStrips) {
  const int lane  = threadIdx.x & 31;
  const int strip = blockIdx.x * (blockDim.x >> 5) + (threadIdx.x >> 5);
  if (strip >= numStrips) return;            // wave-uniform
  const int m0 = strip * 16;

  v8f acc[8];
#pragma unroll
  for (int t = 0; t < 8; ++t) acc[t] = (v8f){0.f,0.f,0.f,0.f,0.f,0.f,0.f,0.f};

#pragma unroll
  for (int k0 = 0; k0 < HDIM; k0 += 32) {
    v16bf aA = load_a_f32(agg, HDIM, m0, k0, lane);   // load once
#pragma unroll
    for (int nt = 0; nt < 8; ++nt) {
      v16bf b = load_b_bf16(WtOut, HDIM, nt * 16, k0, lane);
      acc[nt] = __builtin_amdgcn_wmma_f32_16x16x32_bf16(false, aA, false, b, (short)0, acc[nt], false, false);
    }
    v16bf aZ = load_a_f32(z, HDIM, m0, k0, lane);     // load once
#pragma unroll
    for (int nt = 0; nt < 8; ++nt) {
      v16bf b = load_b_bf16(WtRoot, HDIM, nt * 16, k0, lane);
      acc[nt] = __builtin_amdgcn_wmma_f32_16x16x32_bf16(false, aZ, false, b, (short)0, acc[nt], false, false);
    }
  }

  const int mr = m0 + ((lane & 16) ? 8 : 0);
#pragma unroll
  for (int nt = 0; nt < 8; ++nt) {
    const int n = nt * 16 + (lane & 15);
    const float bv = bias[n];
#pragma unroll
    for (int r = 0; r < 8; ++r) {
      float v = acc[nt][r] + bv;
      zout[(size_t)(mr + r) * HDIM + n] = v > 0.f ? v : 0.f;
    }
  }
}

// h = relu([z[:B] | z[B:2B]] @ W1 + b1); same strip structure, K=256 with the
// concat realized by a K-dependent base pointer (wave-uniform select).
__global__ void k_mlp1(const float* __restrict__ z, const __bf16* __restrict__ Wt1,
                       const float* __restrict__ b1, float* __restrict__ h,
                       int B, int numStrips) {
  const int lane  = threadIdx.x & 31;
  const int strip = blockIdx.x * (blockDim.x >> 5) + (threadIdx.x >> 5);
  if (strip >= numStrips) return;
  const int m0 = strip * 16;

  v8f acc[8];
#pragma unroll
  for (int t = 0; t < 8; ++t) acc[t] = (v8f){0.f,0.f,0.f,0.f,0.f,0.f,0.f,0.f};

#pragma unroll
  for (int k0 = 0; k0 < 2 * HDIM; k0 += 32) {
    const float* Abase = (k0 < HDIM) ? z : (z + (size_t)B * HDIM);
    const int kk = (k0 < HDIM) ? k0 : (k0 - HDIM);
    v16bf a = load_a_f32(Abase, HDIM, m0, kk, lane);
#pragma unroll
    for (int nt = 0; nt < 8; ++nt) {
      v16bf b = load_b_bf16(Wt1, 2 * HDIM, nt * 16, k0, lane);
      acc[nt] = __builtin_amdgcn_wmma_f32_16x16x32_bf16(false, a, false, b, (short)0, acc[nt], false, false);
    }
  }

  const int mr = m0 + ((lane & 16) ? 8 : 0);
#pragma unroll
  for (int nt = 0; nt < 8; ++nt) {
    const int n = nt * 16 + (lane & 15);
    const float bv = b1[n];
#pragma unroll
    for (int r = 0; r < 8; ++r) {
      float v = acc[nt][r] + bv;
      h[(size_t)(mr + r) * HDIM + n] = v > 0.f ? v : 0.f;
    }
  }
}

// pred[m] = h[m] . w2 + b2 ; out[B+m] = (float)label[m]. Wave32 reduction.
__global__ void k_pred(const float* __restrict__ h, const float* __restrict__ w2,
                       const float* __restrict__ b2, const int* __restrict__ lab,
                       float* __restrict__ out, int B) {
  int lane = threadIdx.x & 31;
  int m = blockIdx.x * (blockDim.x >> 5) + (threadIdx.x >> 5);
  if (m >= B) return;
  const float* hp = h + (size_t)m * HDIM;
  float s = 0.f;
#pragma unroll
  for (int i = 0; i < HDIM; i += 32) s += hp[i + lane] * w2[i + lane];
#pragma unroll
  for (int off = 16; off > 0; off >>= 1) s += __shfl_xor(s, off, 32);
  if (lane == 0) {
    out[m] = s + b2[0];
    out[B + m] = (float)lab[m];
  }
}

// ---------------- host launcher ----------------
extern "C" void kernel_launch(void* const* d_in, const int* in_sizes, int n_in,
                              void* d_out, int out_size, void* d_ws, size_t ws_size,
                              hipStream_t stream) {
  const int N = in_sizes[0];           // 50000
  const int E = in_sizes[1];           // 800000
  const int B = in_sizes[3];           // 8192
  const int H = HDIM;                  // 128
  const int L = 2;

  const int*   x          = (const int*)d_in[0];
  const int*   edge_index = (const int*)d_in[2];
  const int*   row        = edge_index;
  const int*   col        = edge_index + E;
  const int*   labels     = (const int*)d_in[3];
  const float* node_emb   = (const float*)d_in[4];
  const float* conv_out_w = (const float*)d_in[6];
  const float* conv_out_b = (const float*)d_in[7];
  const float* conv_root_w= (const float*)d_in[8];
  const float* mlp_w1     = (const float*)d_in[9];
  const float* mlp_b1     = (const float*)d_in[10];
  const float* mlp_w2     = (const float*)d_in[11];
  const float* mlp_b2     = (const float*)d_in[12];
  float* out = (float*)d_out;

  // workspace carve-up (256B aligned)
  char* w = (char*)d_ws;
  size_t off = 0;
  auto take = [&](size_t bytes) { void* p = w + off; off = (off + bytes + 255) & ~(size_t)255; return p; };
  float*  deg  = (float*)take((size_t)N * 4);
  float*  z0   = (float*)take((size_t)N * H * 4);
  float*  z1   = (float*)take((size_t)N * H * 4);
  float*  agg  = (float*)take((size_t)N * H * 4);       // reused as 'h' at the end
  __bf16* wtO0 = (__bf16*)take((size_t)H * H * 2);
  __bf16* wtO1 = (__bf16*)take((size_t)H * H * 2);
  __bf16* wtR0 = (__bf16*)take((size_t)H * H * 2);
  __bf16* wtR1 = (__bf16*)take((size_t)H * H * 2);
  __bf16* wt1  = (__bf16*)take((size_t)2 * H * H * 2);
  __bf16* wtOut[2]  = {wtO0, wtO1};
  __bf16* wtRoot[2] = {wtR0, wtR1};

  const int T = 256;
  // degree -> deg_inv (shared by both layers)
  k_zero<<<(N + T - 1) / T, T, 0, stream>>>(deg, N);
  k_count_deg<<<(E + T - 1) / T, T, 0, stream>>>(row, col, deg, E);
  k_deg_inv<<<(N + T - 1) / T, T, 0, stream>>>(deg, N);
  // embedding gather
  {
    long long tot = (long long)N * H;
    k_embed<<<(unsigned)((tot + T - 1) / T), T, 0, stream>>>(x, node_emb, z0, N);
  }
  // transpose + bf16-convert weights
  for (int l = 0; l < L; ++l) {
    k_wt_bf16<<<(H * H + T - 1) / T, T, 0, stream>>>(conv_out_w  + (size_t)l * H * H, wtOut[l],  H, H);
    k_wt_bf16<<<(H * H + T - 1) / T, T, 0, stream>>>(conv_root_w + (size_t)l * H * H, wtRoot[l], H, H);
  }
  k_wt_bf16<<<(2 * H * H + T - 1) / T, T, 0, stream>>>(mlp_w1, wt1, 2 * H, H);

  // GCN layers
  float* zc = z0;
  float* zn = z1;
  const int gcnStrips = N / 16;                          // 3125
  for (int l = 0; l < L; ++l) {
    long long tot = (long long)N * H;
    k_init_agg<<<(unsigned)((tot + T - 1) / T), T, 0, stream>>>(deg, zc, agg, N);
    k_scatter<<<E, HDIM, 0, stream>>>(row, col, deg, zc, agg);
    k_gcn_gemm<<<(gcnStrips + 7) / 8, T, 0, stream>>>(agg, zc, wtOut[l], wtRoot[l],
                                                      conv_out_b + (size_t)l * H, zn, gcnStrips);
    float* t = zc; zc = zn; zn = t;
  }

  // MLP head (h aliases agg, which is dead after the last GEMM)
  float* h = agg;
  const int mlpStrips = B / 16;                          // 512
  k_mlp1<<<(mlpStrips + 7) / 8, T, 0, stream>>>(zc, wt1, mlp_b1, h, B, mlpStrips);
  k_pred<<<(B + 7) / 8, T, 0, stream>>>(h, mlp_w2, mlp_b2, labels, out, B);
}
